// SDT_18683107738106
// MI455X (gfx1250) — compile-verified, matrix operand order checked by reading, not run on hardware
//
#include <hip/hip_runtime.h>
#include <math.h>

typedef __attribute__((ext_vector_type(16))) __bf16 v16bf;
typedef __attribute__((ext_vector_type(8)))  __bf16 v8bf;
typedef __attribute__((ext_vector_type(4)))  __bf16 v4bf;
typedef __attribute__((ext_vector_type(8)))  float  v8f;

#define B_SZ     16384
#define IN_DIM   2048
#define OUT_DIM  512
#define TDEPTH   10
#define N_NODES  1023
#define N_LEAVES 1024
#define NPAD     1024    // padded node stride for P / Wt rows

// --- CDNA5 async global->LDS copy (ASYNCcnt path) ------------------------
__device__ __forceinline__ unsigned lds_addr32(const void* p) {
    // flat LDS addresses truncate to the LDS byte offset in the low 32 bits
    return (unsigned)(uintptr_t)p;
}
__device__ __forceinline__ void async_copy_b128(unsigned lds, const void* gptr) {
    asm volatile("global_load_async_to_lds_b128 %0, %1, off"
                 :: "v"(lds), "v"(gptr) : "memory");
}
__device__ __forceinline__ void async_wait0() {
    asm volatile("s_wait_asynccnt 0" ::: "memory");
}

// ---------------------------------------------------------------------------
// x: f32 -> bf16 (flat)
// ---------------------------------------------------------------------------
__global__ __launch_bounds__(256)
void sdt_cvt_x(const float* __restrict__ v, __bf16* __restrict__ o)
{
    size_t idx = ((size_t)blockIdx.x * 256 + threadIdx.x) * 4;
    float4 f = *(const float4*)&v[idx];
    v4bf h;
    h[0] = (__bf16)f.x; h[1] = (__bf16)f.y; h[2] = (__bf16)f.z; h[3] = (__bf16)f.w;
    *(v4bf*)&o[idx] = h;
}

// ---------------------------------------------------------------------------
// W (2048 x 1023 f32) -> Wt (1024 x 2048 bf16), transposed, row 1023 zeroed
// ---------------------------------------------------------------------------
__global__ __launch_bounds__(256)
void sdt_cvt_WT(const float* __restrict__ W, __bf16* __restrict__ Wt)
{
    __shared__ float tl[32][33];
    const int t = threadIdx.x;
    const int kt = blockIdx.x * 32;   // K tile
    const int nt = blockIdx.y * 32;   // N tile
    #pragma unroll
    for (int j = 0; j < 4; ++j) {
        int idx = t + 256 * j;
        int kl = idx >> 5, nl = idx & 31;
        int col = nt + nl;
        tl[kl][nl] = (col < N_NODES) ? W[(size_t)(kt + kl) * N_NODES + col] : 0.0f;
    }
    __syncthreads();
    #pragma unroll
    for (int j = 0; j < 4; ++j) {
        int idx = t + 256 * j;
        int nl = idx >> 5, kl = idx & 31;
        Wt[(size_t)(nt + nl) * IN_DIM + kt + kl] = (__bf16)tl[kl][nl];
    }
}

// ---------------------------------------------------------------------------
// value (1024 x 512 f32) -> valT (512 x 1024 bf16), transposed
// ---------------------------------------------------------------------------
__global__ __launch_bounds__(256)
void sdt_cvt_valT(const float* __restrict__ V, __bf16* __restrict__ Vt)
{
    __shared__ float tl[32][33];
    const int t = threadIdx.x;
    const int kt = blockIdx.x * 32;
    const int nt = blockIdx.y * 32;
    #pragma unroll
    for (int j = 0; j < 4; ++j) {
        int idx = t + 256 * j;
        int kl = idx >> 5, nl = idx & 31;
        tl[kl][nl] = V[(size_t)(kt + kl) * OUT_DIM + nt + nl];
    }
    __syncthreads();
    #pragma unroll
    for (int j = 0; j < 4; ++j) {
        int idx = t + 256 * j;
        int nl = idx >> 5, kl = idx & 31;
        Vt[(size_t)(nt + nl) * N_LEAVES + kt + kl] = (__bf16)tl[kl][nl];
    }
}

// ---------------------------------------------------------------------------
// Unified bf16 WMMA GEMM:  C(MxN) = A(MxK) * Bt(NxK)^T
// Block tile 128x128, 8 waves (4 along M x 2 along N), wave tile 32x64.
// Double-buffered LDS tiles filled with global_load_async_to_lds_b128:
// per iteration issue next tile async, compute current, one barrier.
// SIG=true: fused fast-math sigmoid epilogue + reg reduction (GEMM1).
// ---------------------------------------------------------------------------
template<int KDIM, int LDC, bool SIG>
__global__ __launch_bounds__(256)
void sdt_gemm_bf16(const __bf16* __restrict__ A, const __bf16* __restrict__ Bt,
                   const float* __restrict__ bias, float* __restrict__ C,
                   float* __restrict__ reg_out)
{
    __shared__ __align__(16) __bf16 la[2][128][40];  // A tiles (M x K), padded
    __shared__ __align__(16) __bf16 lb[2][128][40];  // B tiles (N x K), padded
    __shared__ float red[256];

    const int t    = threadIdx.x;
    const int lane = t & 31;
    const int wave = t >> 5;
    const int wm   = wave & 3;     // 4 waves along M, 32 rows each
    const int wn   = wave >> 2;    // 2 waves along N, 64 cols each
    const int kg   = lane >> 4;
    const int l15  = lane & 15;
    const int mbase = blockIdx.x * 128;
    const int nbase = blockIdx.y * 128;

    // per-thread async-copy coordinates (chunk c = t, t+256 of 512 16B chunks)
    const int ar0 = t >> 2;                 // rows 0..63 / 64..127
    const int ak  = (t & 3) * 8;

    v8f acc[2][4] = {};

    // prologue: fill buffer 0
    #pragma unroll
    for (int j = 0; j < 2; ++j) {
        int row = ar0 + 64 * j;
        async_copy_b128(lds_addr32(&la[0][row][ak]),
                        &A[(size_t)(mbase + row) * KDIM + ak]);
        async_copy_b128(lds_addr32(&lb[0][row][ak]),
                        &Bt[(size_t)(nbase + row) * KDIM + ak]);
    }

    int cur = 0;
    for (int kb = 0; kb < KDIM; kb += 32) {
        async_wait0();
        __syncthreads();
        int nxt = cur ^ 1;
        if (kb + 32 < KDIM) {
            #pragma unroll
            for (int j = 0; j < 2; ++j) {
                int row = ar0 + 64 * j;
                async_copy_b128(lds_addr32(&la[nxt][row][ak]),
                                &A[(size_t)(mbase + row) * KDIM + kb + 32 + ak]);
                async_copy_b128(lds_addr32(&lb[nxt][row][ak]),
                                &Bt[(size_t)(nbase + row) * KDIM + kb + 32 + ak]);
            }
        }

        v16bf afr[2];
        #pragma unroll
        for (int i = 0; i < 2; ++i) {
            int row = wm * 32 + i * 16 + l15;
            union { v16bf v; v8bf h[2]; } u;
            u.h[0] = *(const v8bf*)&la[cur][row][kg * 8];
            u.h[1] = *(const v8bf*)&la[cur][row][16 + kg * 8];
            afr[i] = u.v;
        }
        #pragma unroll
        for (int j = 0; j < 4; ++j) {
            int n = wn * 64 + j * 16 + l15;
            union { v16bf v; v8bf h[2]; } u;
            u.h[0] = *(const v8bf*)&lb[cur][n][kg * 8];
            u.h[1] = *(const v8bf*)&lb[cur][n][16 + kg * 8];
            v16bf bfr = u.v;
            acc[0][j] = __builtin_amdgcn_wmma_f32_16x16x32_bf16(
                false, afr[0], false, bfr, (short)0, acc[0][j], false, false);
            acc[1][j] = __builtin_amdgcn_wmma_f32_16x16x32_bf16(
                false, afr[1], false, bfr, (short)0, acc[1][j], false, false);
        }
        cur = nxt;
    }
    __syncthreads();   // all reads done before epilogue reuses nothing; keeps waves aligned

    const int mr0 = (lane < 16) ? 0 : 8;
    float local = 0.0f;
    #pragma unroll
    for (int i = 0; i < 2; ++i) {
        #pragma unroll
        for (int j = 0; j < 4; ++j) {
            int n = nbase + wn * 64 + j * 16 + l15;
            float bv = 0.0f;
            if (SIG) bv = (n < N_NODES) ? bias[n] : 0.0f;
            #pragma unroll
            for (int r = 0; r < 8; ++r) {
                int m = mbase + wm * 32 + i * 16 + mr0 + r;
                if (SIG) {
                    float z = acc[i][j][r] + bv;
                    // fast sigmoid: v_exp_f32 + v_rcp_f32
                    float p = __builtin_amdgcn_rcpf(1.0f + __expf(-z));
                    C[(size_t)m * LDC + n] = p;
                    if (n < N_NODES) {
                        int d = 31 - __clz(n + 1);
                        float wgt = -ldexpf(1.0f / (float)B_SZ, -(d + 1)); // -0.5*2^-d/B
                        local += wgt * __logf(fmaxf(p * (1.0f - p), 1e-5f));
                    }
                } else {
                    C[(size_t)m * LDC + n] = acc[i][j][r];
                }
            }
        }
    }
    if (SIG) {
        red[t] = local;
        __syncthreads();
        #pragma unroll
        for (int s = 128; s > 0; s >>= 1) {
            if (t < s) red[t] += red[t + s];
            __syncthreads();
        }
        if (t == 0) atomicAdd(reg_out, red[0]);
    }
}

// ---------------------------------------------------------------------------
// Tree expansion: one block per batch row; level-doubling products in LDS.
// ---------------------------------------------------------------------------
__global__ __launch_bounds__(256)
void sdt_tree_expand(const float* __restrict__ P, __bf16* __restrict__ probs)
{
    __shared__ float Prow[1024];
    __shared__ float cur[1024];
    const int t = threadIdx.x;
    const int row = blockIdx.x;

    #pragma unroll
    for (int j = 0; j < 4; ++j) {
        int i = t + 256 * j;
        Prow[i] = (i < N_NODES) ? P[(size_t)row * NPAD + i] : 0.0f;
    }
    if (t == 0) cur[0] = 1.0f;
    __syncthreads();

    for (int d = 0; d < TDEPTH; ++d) {
        int nlev = 1 << d;
        float o[2], pp[2];
        int cnt = 0;
        for (int i = t; i < nlev; i += 256) { o[cnt] = cur[i]; pp[cnt] = Prow[(nlev - 1) + i]; ++cnt; }
        __syncthreads();
        cnt = 0;
        for (int i = t; i < nlev; i += 256) {
            cur[2 * i]     = o[cnt] * pp[cnt];
            cur[2 * i + 1] = o[cnt] * (1.0f - pp[cnt]);
            ++cnt;
        }
        __syncthreads();
    }
    #pragma unroll
    for (int j = 0; j < 4; ++j) {
        int i = t + 256 * j;
        probs[(size_t)row * N_LEAVES + i] = (__bf16)cur[i];
    }
}

// ---------------------------------------------------------------------------
extern "C" void kernel_launch(void* const* d_in, const int* in_sizes, int n_in,
                              void* d_out, int out_size, void* d_ws, size_t ws_size,
                              hipStream_t stream)
{
    const float* x     = (const float*)d_in[0];   // (16384, 2048)
    const float* W     = (const float*)d_in[1];   // (2048, 1023)
    const float* bias  = (const float*)d_in[2];   // (1023,)
    const float* value = (const float*)d_in[3];   // (1024, 512)

    float* out = (float*)d_out;                   // (16384*512) then reg scalar
    float* reg = out + (size_t)B_SZ * OUT_DIM;

    char* ws = (char*)d_ws;
    size_t off = 0;
    __bf16* xbf   = (__bf16*)(ws + off); off += (size_t)B_SZ * IN_DIM * 2;      // 64 MB
    float*  P     = (float*)(ws + off);  off += (size_t)B_SZ * NPAD * 4;        // 64 MB
    __bf16* probs = (__bf16*)(ws + off); off += (size_t)B_SZ * N_LEAVES * 2;    // 32 MB
    __bf16* WtBF  = (__bf16*)(ws + off); off += (size_t)NPAD * IN_DIM * 2;      //  4 MB
    __bf16* valT  = (__bf16*)(ws + off); off += (size_t)OUT_DIM * N_LEAVES * 2; //  1 MB

    hipMemsetAsync(reg, 0, sizeof(float), stream);

    sdt_cvt_x<<<(size_t)B_SZ * IN_DIM / (256 * 4), 256, 0, stream>>>(x, xbf);
    sdt_cvt_WT<<<dim3(IN_DIM / 32, NPAD / 32), 256, 0, stream>>>(W, WtBF);
    sdt_cvt_valT<<<dim3(N_LEAVES / 32, OUT_DIM / 32), 256, 0, stream>>>(value, valT);

    sdt_gemm_bf16<IN_DIM, NPAD, true>
        <<<dim3(B_SZ / 128, NPAD / 128), 256, 0, stream>>>(xbf, WtBF, bias, P, reg);

    sdt_tree_expand<<<B_SZ, 256, 0, stream>>>(P, probs);

    sdt_gemm_bf16<N_LEAVES, OUT_DIM, false>
        <<<dim3(B_SZ / 128, OUT_DIM / 128), 256, 0, stream>>>(probs, valT, bias, out, reg);
}
